// Net_12481174962491
// MI455X (gfx1250) — matrix operand order repaired and checked by reading, not compiled, not run on hardware
//
#include <hip/hip_runtime.h>
#include <hip/hip_bf16.h>

// ---------------------------------------------------------------------------
// CDNA5 (gfx1250) implementation of the reference CNN.
// Convs = implicit GEMM on v_wmma_f32_16x16x32_f16 (f16 operands staged in
// LDS, f32 accumulation). Weight tiles are DMA'd by the Tensor Data Mover
// (tensor_load_to_lds + s_wait_tensorcnt); activations/weights are kept as
// f16 shadows in memory to halve gather bandwidth.
// Block tile 32(M) x 256(N): each wave32 computes a 32x32 output patch with
// 4 WMMAs per 32-wide K step (2 A-subtiles x 2 B-subtiles).
// ---------------------------------------------------------------------------

typedef __attribute__((ext_vector_type(16))) _Float16     v16h;
typedef __attribute__((ext_vector_type(8)))  float        v8f;
typedef __attribute__((ext_vector_type(4)))  unsigned int uint32x4;
typedef __attribute__((ext_vector_type(8)))  unsigned int uint32x8;

#define KTAB_MAX 1280   // >= max Cin*KH*KW (res blocks: 128*9 = 1152)

__global__ __launch_bounds__(256) void conv_implicit_gemm_wmma(
    const _Float16* __restrict__ Xh, const _Float16* __restrict__ Wh,
    const float* __restrict__ Bias, float* __restrict__ Y, _Float16* __restrict__ Yh,
    int Cin, int Hin, int Win,
    int Cout, int Hout, int Wout,
    int KH, int KW, int stride, int pad, int relu,
    int wShift, int hShift, int Npos, int Ktot)
{
    __shared__ _Float16 sA[32 * 32];     // 32 Cout x 32 K (TDM tile)
    __shared__ _Float16 sB[256 * 32];    // 256 N  x 32 K (N-major)
    __shared__ unsigned ktab[KTAB_MAX];  // k -> (kh<<28)|(kw<<24)|(ci*Hin*Win)

    const int tid  = threadIdx.x;
    const int lane = tid & 31;
    const int wave = tid >> 5;
    const int nBase  = blockIdx.x * 256;
    const int coBase = blockIdx.y * 32;
    const int HinWin = Hin * Win;

    // ---- once per block: k-decomposition table ----------------------------
    for (int k = tid; k < Ktot; k += 256) {
        int s = k % KW; int u = k / KW;
        int r = u % KH; int ci = u / KH;
        ktab[k] = ((unsigned)r << 28) | ((unsigned)s << 24) | (unsigned)(ci * HinWin);
    }

    // ---- per-thread N column (fixed across the whole K loop) --------------
    const int gn = nBase + tid;
    const bool nv = gn < Npos;
    int ow = 0, oh = 0, bImg = 0;
    if (nv) {
        ow   = gn & (Wout - 1);
        oh   = (gn >> wShift) & (Hout - 1);
        bImg = gn >> (wShift + hShift);
    }
    const size_t xBase = (size_t)bImg * Cin * HinWin;
    const int ihBase = oh * stride - pad;
    const int iwBase = ow * stride - pad;

    v8f acc00 = {0.f,0.f,0.f,0.f,0.f,0.f,0.f,0.f};   // M 0-15,  N 0-15
    v8f acc01 = {0.f,0.f,0.f,0.f,0.f,0.f,0.f,0.f};   // M 0-15,  N 16-31
    v8f acc10 = {0.f,0.f,0.f,0.f,0.f,0.f,0.f,0.f};   // M 16-31, N 0-15
    v8f acc11 = {0.f,0.f,0.f,0.f,0.f,0.f,0.f,0.f};   // M 16-31, N 16-31

    __syncthreads();  // ktab ready

    for (int kb = 0; kb < Ktot; kb += 32) {
        // ---- A tile via Tensor Data Mover (one wave issues the DMA) -------
        if (tid == 0) {
            unsigned ldsOff = (unsigned)(size_t)(&sA[0]);   // LDS byte offset
            unsigned long long ga =
                (unsigned long long)(const void*)(Wh + (size_t)coBase * Ktot + kb);
            unsigned td0 = (unsigned)(Ktot - kb);           // remaining K   (OOB -> 0)
            unsigned td1 = (unsigned)(Cout - coBase);       // remaining rows (OOB -> 0)
            uint32x4 g0;
            g0[0] = 1u;                                     // count=1 (valid), user mode
            g0[1] = ldsOff;                                 // lds_addr
            g0[2] = (unsigned)(ga & 0xFFFFFFFFu);           // global_addr lo
            g0[3] = (unsigned)((ga >> 32) & 0x1FFFFFFu) | (2u << 30);  // hi | type=2
            uint32x8 g1;
            g1[0] = 1u << 16;                               // workgroup_mask=0, data_size=2B
            g1[1] = (td0 & 0xFFFFu) << 16;                  // tensor_dim0 lo16
            g1[2] = ((td0 >> 16) & 0xFFFFu) | ((td1 & 0xFFFFu) << 16);
            g1[3] = ((td1 >> 16) & 0xFFFFu) | (32u << 16);  // tile_dim0 = 32 (K)
            g1[4] = 32u;                                    // tile_dim1 = 32 (rows)
            g1[5] = (unsigned)Ktot;                         // tensor_dim0_stride (row pitch)
            g1[6] = 0u;
            g1[7] = 0u;
            asm volatile("tensor_load_to_lds %0, %1" :: "s"(g0), "s"(g1) : "memory");
            __builtin_amdgcn_s_wait_tensorcnt(0);
        }
        if (kb + 32 < Ktot)  // next weight chunk into cache (global_prefetch_b8)
            __builtin_prefetch(&Wh[(size_t)coBase * Ktot + kb + 32], 0, 1);

        // ---- B tile: im2col gather, one N column per thread ---------------
        // Phase 1: batched loads into registers (many outstanding loads);
        // Phase 2: contiguous LDS stores (vectorizable to b128).
        {
            _Float16 tmp[32];
#pragma unroll
            for (int kk = 0; kk < 32; ++kk) {
                int k = kb + kk;
                _Float16 v = (_Float16)0.0f;
                if (nv && k < Ktot) {
                    unsigned e = ktab[k];
                    int ih = ihBase + (int)(e >> 28);
                    int iw = iwBase + (int)((e >> 24) & 15u);
                    if ((unsigned)ih < (unsigned)Hin && (unsigned)iw < (unsigned)Win)
                        v = Xh[xBase + (e & 0xFFFFFFu) + (size_t)ih * Win + iw];
                }
                tmp[kk] = v;
            }
            _Float16* dst = &sB[tid * 32];
#pragma unroll
            for (int kk = 0; kk < 32; ++kk) dst[kk] = tmp[kk];
        }
        __syncthreads();

        // ---- fragments + 4 WMMAs ------------------------------------------
        // A 16x32 f16: lanes 0-15 M=lane, K={0..7,16..23}; lanes 16-31 K={8..15,24..31}
        const int mrow = lane & 15;
        const int hiK  = (lane >> 4) << 3;
        const _Float16* pa0 = &sA[mrow * 32];
        const _Float16* pa1 = &sA[(16 + mrow) * 32];
        v16h a0, a1, b0, b1;
#pragma unroll
        for (int i = 0; i < 8; ++i) {
            a0[i]     = pa0[hiK + i];
            a0[8 + i] = pa0[16 + hiK + i];
            a1[i]     = pa1[hiK + i];
            a1[8 + i] = pa1[16 + hiK + i];
        }
        // B 32x16 f16: lanes 0-15 N=lane K={0..7,16..23}; lanes 16-31 N=lane-16 K={8..15,24..31}
        const _Float16* pb = &sB[(wave * 32 + (lane & 15)) * 32];
#pragma unroll
        for (int i = 0; i < 8; ++i) {
            b0[i] = pb[hiK + i]; b0[8 + i] = pb[16 + hiK + i];
            b1[i] = pb[16 * 32 + hiK + i]; b1[8 + i] = pb[16 * 32 + 16 + hiK + i];
        }

        acc00 = __builtin_amdgcn_wmma_f32_16x16x32_f16(false, a0, false, b0,
                                                       (short)0, acc00, false, false);
        acc01 = __builtin_amdgcn_wmma_f32_16x16x32_f16(false, a0, false, b1,
                                                       (short)0, acc01, false, false);
        acc10 = __builtin_amdgcn_wmma_f32_16x16x32_f16(false, a1, false, b0,
                                                       (short)0, acc10, false, false);
        acc11 = __builtin_amdgcn_wmma_f32_16x16x32_f16(false, a1, false, b1,
                                                       (short)0, acc11, false, false);
        __syncthreads();
    }

    // ---- store D: vgpr i -> M = i + (lane>=16 ? 8 : 0), N per lane --------
    const int mhi = (lane >> 4) << 3;
    const int nb  = nBase + wave * 32 + (lane & 15);
#pragma unroll
    for (int t = 0; t < 2; ++t) {
        int n = nb + 16 * t;
        if (n < Npos) {
            int sow = n & (Wout - 1);
            int soh = (n >> wShift) & (Hout - 1);
            int sb  = n >> (wShift + hShift);
#pragma unroll
            for (int ms = 0; ms < 2; ++ms) {
                const v8f& acc = ms ? (t ? acc11 : acc10) : (t ? acc01 : acc00);
#pragma unroll
                for (int i = 0; i < 8; ++i) {
                    int co = coBase + ms * 16 + mhi + i;
                    if (co < Cout) {
                        float v = acc[i] + Bias[co];
                        if (relu) v = fmaxf(v, 0.0f);
                        size_t idx = (((size_t)sb * Cout + co) << (wShift + hShift)) +
                                     ((size_t)soh << wShift) + sow;
                        Y[idx] = v;
                        if (Yh) Yh[idx] = (_Float16)v;
                    }
                }
            }
        }
    }
}

// ---------------------------------------------------------------------------
// Training-mode BatchNorm statistics + apply (optional f16 shadow output).
// ---------------------------------------------------------------------------
__global__ __launch_bounds__(256) void bn_stats_kernel(
    const float* __restrict__ X, float* __restrict__ mean, float* __restrict__ rstd,
    int Bn, int C, int HW, float eps)
{
    __shared__ float ssum[256];
    __shared__ float ssq[256];
    const int c = blockIdx.x;
    const int total = Bn * HW;
    float s = 0.f, q = 0.f;
    for (int i = threadIdx.x; i < total; i += 256) {
        int b = i / HW, p = i - b * HW;
        float v = X[((size_t)b * C + c) * HW + p];
        s += v; q += v * v;
    }
    ssum[threadIdx.x] = s; ssq[threadIdx.x] = q;
    __syncthreads();
    for (int off = 128; off > 0; off >>= 1) {
        if (threadIdx.x < off) {
            ssum[threadIdx.x] += ssum[threadIdx.x + off];
            ssq[threadIdx.x]  += ssq[threadIdx.x + off];
        }
        __syncthreads();
    }
    if (threadIdx.x == 0) {
        float inv = 1.0f / (float)total;
        float mu  = ssum[0] * inv;
        float var = ssq[0] * inv - mu * mu;
        mean[c] = mu;
        rstd[c] = rsqrtf(var + eps);
    }
}

__global__ void bn_apply_kernel(float* __restrict__ X, _Float16* __restrict__ Xh,
                                const float* __restrict__ mean, const float* __restrict__ rstd,
                                const float* __restrict__ g, const float* __restrict__ be,
                                int C, int HW, int total, int relu)
{
    int i = blockIdx.x * blockDim.x + threadIdx.x;
    if (i >= total) return;
    int c = (i / HW) % C;
    float v = (X[i] - mean[c]) * rstd[c] * g[c] + be[c];
    if (relu) v = fmaxf(v, 0.0f);
    X[i] = v;
    if (Xh) Xh[i] = (_Float16)v;
}

// cat(f16) = [blur(64ch broadcast of conv7 8x8), src(64ch)] @64x64, B=32.
__global__ void concat_blur_kernel(const float* __restrict__ blur64,
                                   const float* __restrict__ src,
                                   _Float16* __restrict__ dst, int total)
{
    int i = blockIdx.x * blockDim.x + threadIdx.x;
    if (i >= total) return;
    int p  = i & 4095;
    int ch = (i >> 12) & 127;
    int b  = i >> 19;
    float v = (ch < 64) ? blur64[(b << 6) + ch]
                        : src[((((size_t)b << 6) + (ch - 64)) << 12) + p];
    dst[i] = (_Float16)v;
}

__global__ void add3_kernel(const float* __restrict__ a, const float* __restrict__ b,
                            const float* __restrict__ c, float* __restrict__ o, int total)
{
    int i = blockIdx.x * blockDim.x + threadIdx.x;
    if (i >= total) return;
    o[i] = a[i] + b[i] + c[i];
}

__global__ void cvt_f32_f16_kernel(const float* __restrict__ X, _Float16* __restrict__ Y, int total)
{
    int i = blockIdx.x * blockDim.x + threadIdx.x;
    if (i >= total) return;
    Y[i] = (_Float16)X[i];
}

// PixelShuffle(4) [B,16,64,64] -> [B,1,256,256], fused sigmoid.
__global__ void shuffle_sigmoid_kernel(const float* __restrict__ X, float* __restrict__ O, int total)
{
    int i = blockIdx.x * blockDim.x + threadIdx.x;
    if (i >= total) return;
    int w = i & 255; int t = i >> 8;
    int h = t & 255; int b = t >> 8;
    int ch = (h & 3) * 4 + (w & 3);
    float v = X[((((size_t)b * 16 + ch) * 64) + (h >> 2)) * 64 + (w >> 2)];
    O[i] = 1.0f / (1.0f + expf(-v));
}

// ---------------------------------------------------------------------------
// Orchestration. Input order (setup_inputs dict order, params insertion
// order): d_in[0]=x_1, d_in[1]=epoch, d_in[2+2i]/[3+2i]=conv i w/b (20),
// d_in[42+2j]/[43+2j]=bn j gamma/beta (7).
// ---------------------------------------------------------------------------
extern "C" void kernel_launch(void* const* d_in, const int* in_sizes, int n_in,
                              void* d_out, int out_size, void* d_ws, size_t ws_size,
                              hipStream_t stream)
{
    (void)in_sizes; (void)n_in; (void)out_size; (void)ws_size;
    const float* x1 = (const float*)d_in[0];
    auto cw = [&](int i) { return (const float*)d_in[2 + 2 * i]; };
    auto cb = [&](int i) { return (const float*)d_in[2 + 2 * i + 1]; };
    auto bg = [&](int j) { return (const float*)d_in[42 + 2 * j]; };
    auto bb = [&](int j) { return (const float*)d_in[42 + 2 * j + 1]; };

    char* wsp = (char*)d_ws;
    auto alloc = [&](size_t elems) -> float* {
        float* p = (float*)wsp;
        wsp += ((elems * sizeof(float)) + 255) & ~(size_t)255;
        return p;
    };
    auto allocH = [&](size_t elems) -> _Float16* {
        _Float16* p = (_Float16*)wsp;
        wsp += ((elems * sizeof(_Float16)) + 255) & ~(size_t)255;
        return p;
    };

    const int EB = 256;

    // ---- f16 weight shadows ----------------------------------------------
    static const int shape[20][3] = {
        {16,1,3},{32,16,3},{64,32,5},{4,64,3},{30,4,3},{78,30,5},{8,78,3},{1,8,1},
        {32,128,3},{64,32,3},{64,128,1},{50,128,3},{27,50,3},{64,27,3},{64,128,1},
        {40,128,3},{64,40,3},{64,128,1},{30,128,3},{16,30,1}};
    _Float16* WhArr[20];
    for (int i = 0; i < 20; ++i) {
        int n = shape[i][0] * shape[i][1] * shape[i][2] * shape[i][2];
        WhArr[i] = allocH((size_t)n);
        cvt_f32_f16_kernel<<<(n + EB - 1) / EB, EB, 0, stream>>>(cw(i), WhArr[i], n);
    }

    auto conv = [&](const _Float16* Xh, int widx, float* Y, _Float16* Yh,
                    int Cin, int Hin, int Win, int Cout, int K, int stride,
                    int pad, int relu) {
        int Hout = (Hin + 2 * pad - K) / stride + 1;
        int Wout = (Win + 2 * pad - K) / stride + 1;
        int Npos = 32 * Hout * Wout;
        int Ktot = Cin * K * K;
        int wS = __builtin_ctz(Wout), hS = __builtin_ctz(Hout);
        dim3 grid((Npos + 255) / 256, (Cout + 31) / 32);
        conv_implicit_gemm_wmma<<<grid, 256, 0, stream>>>(
            Xh, WhArr[widx], cb(widx), Y, Yh,
            Cin, Hin, Win, Cout, Hout, Wout, K, K, stride, pad, relu, wS, hS, Npos, Ktot);
    };

    float* mean = alloc(128);
    float* rstd = alloc(128);
    auto bn = [&](float* X, int j, int C, int relu, _Float16* Xh) {
        const int HW = 64 * 64;
        const int total = 32 * C * HW;
        bn_stats_kernel<<<C, 256, 0, stream>>>(X, mean, rstd, 32, C, HW, 1e-5f);
        bn_apply_kernel<<<(total + EB - 1) / EB, EB, 0, stream>>>(
            X, Xh, mean, rstd, bg(j), bb(j), C, HW, total, relu);
    };

    // ---- buffers ----------------------------------------------------------
    const size_t n64 = (size_t)32 * 64 * 64 * 64;   // 64-ch tensor @64x64
    float* t1 = alloc((size_t)32 * 16 * 128 * 128);
    float* t2 = alloc((size_t)32 * 32 * 128 * 128);
    float* ds = alloc(n64);
    float* t3 = alloc((size_t)32 * 4 * 32 * 32);
    float* t4 = alloc((size_t)32 * 30 * 16 * 16);
    float* t5 = alloc((size_t)32 * 78 * 8 * 8);
    float* t6 = alloc((size_t)32 * 8 * 8 * 8);
    float* t7 = alloc((size_t)32 * 64);
    float* ra = alloc((size_t)32 * 50 * 64 * 64);
    float* rb = alloc(n64);
    float* px = alloc(n64);
    float* ro = alloc(n64);

    _Float16* x1h = allocH((size_t)32 * 256 * 256);
    _Float16* t1h = allocH((size_t)32 * 16 * 128 * 128);
    _Float16* t2h = allocH((size_t)32 * 32 * 128 * 128);
    _Float16* dsh = allocH(n64);
    _Float16* t3h = allocH((size_t)32 * 4 * 32 * 32);
    _Float16* t4h = allocH((size_t)32 * 30 * 16 * 16);
    _Float16* t5h = allocH((size_t)32 * 78 * 8 * 8);
    _Float16* t6h = allocH((size_t)32 * 8 * 8 * 8);
    _Float16* cath = allocH((size_t)32 * 128 * 64 * 64);
    _Float16* rah  = allocH((size_t)32 * 50 * 64 * 64);
    _Float16* pxh  = allocH((size_t)32 * 27 * 64 * 64);

    // ---- encoder ----------------------------------------------------------
    {
        int n = 32 * 256 * 256;
        cvt_f32_f16_kernel<<<(n + EB - 1) / EB, EB, 0, stream>>>(x1, x1h, n);
    }
    conv(x1h, 0, t1, t1h, 1, 256, 256, 16, 3, 2, 1, 1);    // conv1 + relu -> 128
    conv(t1h, 1, t2, t2h, 16, 128, 128, 32, 3, 1, 1, 0);   // conv2 (no relu)
    conv(t2h, 2, ds, dsh, 32, 128, 128, 64, 5, 2, 2, 1);   // conv2_1 + relu -> 64
    conv(dsh, 3, t3, t3h, 64, 64, 64, 4, 3, 2, 1, 1);      // conv3 -> 32
    conv(t3h, 4, t4, t4h, 4, 32, 32, 30, 3, 2, 1, 1);      // conv4 -> 16
    conv(t4h, 5, t5, t5h, 30, 16, 16, 78, 5, 2, 2, 1);     // conv5 -> 8
    conv(t5h, 6, t6, t6h, 78, 8, 8, 8, 3, 1, 1, 1);        // conv6
    conv(t6h, 7, t7, nullptr, 8, 8, 8, 1, 1, 1, 0, 1);     // conv7 -> blur source

    // ---- residual blocks at 64x64 ----------------------------------------
    const int tot64  = (int)n64;
    const int totCat = 32 << 19;
    auto cat_of = [&](const float* src) {
        concat_blur_kernel<<<(totCat + EB - 1) / EB, EB, 0, stream>>>(t7, src, cath, totCat);
    };

    // block 1
    cat_of(ds);
    conv(cath, 8, ra, nullptr, 128, 64, 64, 32, 3, 1, 1, 0);  bn(ra, 0, 32, 1, rah);
    conv(rah, 9, rb, nullptr, 32, 64, 64, 64, 3, 1, 1, 0);    bn(rb, 1, 64, 0, nullptr);
    conv(cath, 10, px, nullptr, 128, 64, 64, 64, 1, 1, 0, 0);
    add3_kernel<<<(tot64 + EB - 1) / EB, EB, 0, stream>>>(rb, px, ds, ro, tot64);

    // block 2
    cat_of(ro);
    conv(cath, 11, ra, nullptr, 128, 64, 64, 50, 3, 1, 1, 0); bn(ra, 2, 50, 1, rah);
    conv(rah, 12, px, nullptr, 50, 64, 64, 27, 3, 1, 1, 0);   bn(px, 3, 27, 1, pxh);
    conv(pxh, 13, rb, nullptr, 27, 64, 64, 64, 3, 1, 1, 0);   bn(rb, 4, 64, 0, nullptr);
    conv(cath, 14, px, nullptr, 128, 64, 64, 64, 1, 1, 0, 0);
    add3_kernel<<<(tot64 + EB - 1) / EB, EB, 0, stream>>>(rb, px, ds, ro, tot64);

    // block 3
    cat_of(ro);
    conv(cath, 15, ra, nullptr, 128, 64, 64, 40, 3, 1, 1, 0); bn(ra, 5, 40, 1, rah);
    conv(rah, 16, rb, nullptr, 40, 64, 64, 64, 3, 1, 1, 0);   bn(rb, 6, 64, 0, nullptr);
    conv(cath, 17, px, nullptr, 128, 64, 64, 64, 1, 1, 0, 0);
    add3_kernel<<<(tot64 + EB - 1) / EB, EB, 0, stream>>>(rb, px, ds, ro, tot64);

    // ---- head -------------------------------------------------------------
    cat_of(ro);
    conv(cath, 18, ra, rah, 128, 64, 64, 30, 3, 1, 1, 1);   // up_conv1 + relu (f16 shadow)
    conv(rah, 19, rb, nullptr, 30, 64, 64, 16, 1, 1, 0, 1); // up_conv2 + relu
    const int totOut = 32 * 256 * 256;
    shuffle_sigmoid_kernel<<<(totOut + EB - 1) / EB, EB, 0, stream>>>(rb, (float*)d_out, totOut);
}
